// DifferentiableMemory_20229295964742
// MI455X (gfx1250) — compile-verified, matrix operand order checked by reading, not run on hardware
//
#include <hip/hip_runtime.h>
#include <hip/hip_bf16.h>
#include <stdint.h>

// ---------------------------------------------------------------------------
// Shapes (fixed by the reference): B=8, S=4096, H=1024, M=1024, TEMP=1.0
// ---------------------------------------------------------------------------
#define BB 8
#define SS 4096
#define HH 1024
#define MM 1024

typedef __attribute__((ext_vector_type(8)))  float  v8f;
typedef __attribute__((ext_vector_type(16))) __bf16 v16bf;

#if defined(__gfx1250__) && __has_builtin(__builtin_amdgcn_tensor_load_to_lds) && \
    __has_builtin(__builtin_amdgcn_s_wait_tensorcnt)
#define USE_TDM 1
typedef unsigned int v4u __attribute__((ext_vector_type(4)));
typedef int v8i __attribute__((ext_vector_type(8)));
typedef int v4i __attribute__((ext_vector_type(4)));
#else
#define USE_TDM 0
#endif

__device__ inline v8f zero_v8f() {
  v8f z;
#pragma unroll
  for (int i = 0; i < 8; ++i) z[i] = 0.0f;
  return z;
}

__device__ inline v8f wmma_bf16(v16bf a, v16bf b, v8f c) {
  return __builtin_amdgcn_wmma_f32_16x16x32_bf16(false, a, false, b, (short)0, c,
                                                 false, false);
}

// ---------------------------------------------------------------------------
// f32 -> bf16 flat convert
// ---------------------------------------------------------------------------
__global__ void __launch_bounds__(256) f32_to_bf16_kernel(const float* __restrict__ in,
                                                          __bf16* __restrict__ out,
                                                          size_t n) {
  size_t i = (size_t)blockIdx.x * 256 + threadIdx.x;
  if (i < n) out[i] = (__bf16)in[i];
}

// Split Wo (H x 2H, row-major) into two H x H bf16 halves.
__global__ void __launch_bounds__(256) split_wo_kernel(const float* __restrict__ Wo,
                                                       __bf16* __restrict__ Wo1,
                                                       __bf16* __restrict__ Wo2) {
  size_t i = (size_t)blockIdx.x * 256 + threadIdx.x;  // over H*H
  if (i >= (size_t)HH * HH) return;
  size_t n = i / HH, k = i % HH;
  Wo1[i] = (__bf16)Wo[n * (2 * HH) + k];
  Wo2[i] = (__bf16)Wo[n * (2 * HH) + HH + k];
}

// Gather init_states rows (linspace sampling of S) and convert to bf16.
__global__ void __launch_bounds__(256) gather_bf16_kernel(const float* __restrict__ hs,
                                                          __bf16* __restrict__ Xm) {
  size_t i = (size_t)blockIdx.x * 256 + threadIdx.x;  // over B*M*H
  if (i >= (size_t)BB * MM * HH) return;
  size_t h = i % HH;
  size_t t = i / HH;
  int m = (int)(t % MM);
  int b = (int)(t / MM);
  int src = (int)((float)m * (float)(SS - 1) / (float)(MM - 1));
  if (src > SS - 1) src = SS - 1;
  Xm[i] = (__bf16)hs[((size_t)b * SS + src) * HH + h];
}

// ---------------------------------------------------------------------------
// Row-wise L2 normalize: f32 (rows x H) -> bf16 (rows x H). One block per row.
// ---------------------------------------------------------------------------
__global__ void __launch_bounds__(256) l2norm_bf16_kernel(const float* __restrict__ in,
                                                          __bf16* __restrict__ out) {
  __shared__ float red[256];
  size_t row = blockIdx.x;
  const float* x = in + row * HH;
  float s = 0.0f;
  for (int j = threadIdx.x; j < HH; j += 256) {
    float v = x[j];
    s += v * v;
  }
  red[threadIdx.x] = s;
  __syncthreads();
  for (int off = 128; off > 0; off >>= 1) {
    if (threadIdx.x < (unsigned)off) red[threadIdx.x] += red[threadIdx.x + off];
    __syncthreads();
  }
  float nrm = sqrtf(red[0]);
  float inv = 1.0f / fmaxf(nrm, 1e-12f);
  __bf16* o = out + row * HH;
  for (int j = threadIdx.x; j < HH; j += 256) o[j] = (__bf16)(x[j] * inv);
}

// ---------------------------------------------------------------------------
// Generic batched bf16 WMMA GEMM:  C[z] = A[z] (rows x K) * W[z] (cols x K)^T + bias
// grid = (cols/512, rows/16, batch), block = 256 threads (8 waves).
// Each wave owns 4 accumulator tiles. The k-loop is 2x-unrolled with two
// independent register sets (ping-pong): loads write directly into the VGPR
// tuples the next WMMA consumes, eliminating fragment-copy mov chains while
// keeping a half-iteration of load latency covered by matrix ops.
// Requires K % 64 == 0 (true for all uses here: K = 1024).
// ---------------------------------------------------------------------------
__global__ void __launch_bounds__(256) gemm_bf16_wmma_kernel(
    const __bf16* __restrict__ A, long long sAb, int lda,
    const __bf16* __restrict__ W, long long sWb, int ldw,
    const float* __restrict__ bias,
    float* __restrict__ C, long long sCb, int ldc,
    __bf16* __restrict__ Obf, long long sOb, int ldo, int otrans,
    int K, int accumulate) {
  const int lane = threadIdx.x & 31;
  const int wave = threadIdx.x >> 5;
  const int half = lane >> 4;   // 0: lanes 0-15, 1: lanes 16-31
  const int l16  = lane & 15;
  const int z = blockIdx.z;
  const int rowbase = blockIdx.y * 16;
  const int colbase = blockIdx.x * 512 + wave * 64;

  const __bf16* Ap = A + (size_t)z * sAb + (size_t)(rowbase + l16) * lda + half * 16;
  const __bf16* Wp = W + (size_t)z * sWb + (size_t)(colbase + l16) * ldw + half * 16;
  const size_t cstride = (size_t)16 * ldw;

  v8f acc[4];
#pragma unroll
  for (int c = 0; c < 4; ++c) acc[c] = zero_v8f();

  // Ping-pong register sets.
  v16bf a0 = *(const v16bf*)(Ap);
  v16bf b0[4];
#pragma unroll
  for (int c = 0; c < 4; ++c) b0[c] = *(const v16bf*)(Wp + c * cstride);

  int kk = 0;
  for (; kk + 64 < K; kk += 64) {
    v16bf a1 = *(const v16bf*)(Ap + kk + 32);
    v16bf b1[4];
#pragma unroll
    for (int c = 0; c < 4; ++c) b1[c] = *(const v16bf*)(Wp + c * cstride + kk + 32);
#pragma unroll
    for (int c = 0; c < 4; ++c) acc[c] = wmma_bf16(a0, b0[c], acc[c]);

    a0 = *(const v16bf*)(Ap + kk + 64);
#pragma unroll
    for (int c = 0; c < 4; ++c) b0[c] = *(const v16bf*)(Wp + c * cstride + kk + 64);
#pragma unroll
    for (int c = 0; c < 4; ++c) acc[c] = wmma_bf16(a1, b1[c], acc[c]);
  }
  // Tail: set0 holds fragments for K-64; load K-32 into set1.
  {
    v16bf a1 = *(const v16bf*)(Ap + K - 32);
    v16bf b1[4];
#pragma unroll
    for (int c = 0; c < 4; ++c) b1[c] = *(const v16bf*)(Wp + c * cstride + K - 32);
#pragma unroll
    for (int c = 0; c < 4; ++c) acc[c] = wmma_bf16(a0, b0[c], acc[c]);
#pragma unroll
    for (int c = 0; c < 4; ++c) acc[c] = wmma_bf16(a1, b1[c], acc[c]);
  }

#pragma unroll
  for (int c = 0; c < 4; ++c) {
    int col = colbase + c * 16 + l16;
    float bv = bias ? bias[col] : 0.0f;
#pragma unroll
    for (int r = 0; r < 8; ++r) {
      int row = rowbase + r + 8 * half;
      float v = acc[c][r] + bv;
      if (C) {
        float* cp = C + (size_t)z * sCb + (size_t)row * ldc + col;
        if (accumulate) v += *cp;
        *cp = v;
      }
      if (Obf) {
        if (otrans)
          Obf[(size_t)z * sOb + (size_t)col * ldo + row] = (__bf16)v;
        else
          Obf[(size_t)z * sOb + (size_t)row * ldo + col] = (__bf16)v;
      }
    }
  }
}

// ---------------------------------------------------------------------------
// Fused attention: per (batch, 16-query tile), block = 512 threads (16 waves),
// wave w owns 64 columns (4 tiles) of M in phase A and of H in phase C.
//   scores = qn_tile (16xH) . kn^T (HxM)  [WMMA, scores live in registers]
//   softmax over M (shfl + 16-wave LDS reduce)
//   retrieved = probs (16xM) . mv (MxH)   [WMMA, B fragments from mv^T]
// qn tile staging uses the Tensor Data Mover when available (contiguous 32KB
// tile, D# count=1, data_size=2B), waited with s_wait_tensorcnt.
// Both WMMA loops use the same ping-pong double-buffered register scheme.
// ---------------------------------------------------------------------------
__global__ void __launch_bounds__(512) attn_kernel(const __bf16* __restrict__ qn,
                                                   const __bf16* __restrict__ kn,
                                                   const __bf16* __restrict__ mvT,
                                                   __bf16* __restrict__ Rb) {
  __shared__ __align__(16) __bf16 sQP[16 * 1024];  // qn tile, later reused for probs
  __shared__ float wred[16][16];
  __shared__ float rowstat[16];

  const int tid  = threadIdx.x;
  const int lane = tid & 31;
  const int wave = tid >> 5;       // 0..15
  const int half = lane >> 4;
  const int l16  = lane & 15;
  const int b     = blockIdx.y;
  const int qbase = blockIdx.x * 16;

  // ---- Stage qn tile (16 x 1024 bf16 = 32KB) into LDS ----------------------
#if USE_TDM
  if (tid < 32) {  // one wave issues the DMA; EXEC is ignored by TDM
    uint64_t ga = (uint64_t)(uintptr_t)(qn + ((size_t)b * SS + qbase) * HH);
    uint32_t lds_off = (uint32_t)(uintptr_t)(&sQP[0]);
    v4u g0;
    g0[0] = 1u;                                   // count = 1 valid descriptor
    g0[1] = lds_off;                              // lds_addr (bytes)
    g0[2] = (uint32_t)ga;                         // global_addr[31:0]
    g0[3] = (uint32_t)((ga >> 32) & 0x1FFFFFFu)   // global_addr[56:32]
            | (2u << 30);                         // type = 2 ("image")
    v8i g1;
    g1[0] = 0x00010000;   // workgroup_mask=0, data_size=1 (2 bytes)
    g1[1] = 0x40000000;   // tensor_dim0[15:0]=0x4000 in bits [31:16]
    g1[2] = 0x00010000;   // tensor_dim0[31:16]=0, tensor_dim1[15:0]=1
    g1[3] = 0x40000000;   // tensor_dim1[31:16]=0, tile_dim0=0x4000 (16384 elems)
    g1[4] = 1;            // tile_dim1=1, tile_dim2=0
    g1[5] = 16384;        // tensor_dim0_stride[31:0]
    g1[6] = 0;            // tensor_dim0_stride[47:32]=0, tensor_dim1_stride lo=0
    g1[7] = 0;
    v4i gz;
    gz[0] = 0; gz[1] = 0; gz[2] = 0; gz[3] = 0;
#if __clang_major__ >= 23
    v8i gz8;
#pragma unroll
    for (int i = 0; i < 8; ++i) gz8[i] = 0;
    __builtin_amdgcn_tensor_load_to_lds(g0, g1, gz, gz, gz8, 0);
#else
    __builtin_amdgcn_tensor_load_to_lds(g0, g1, gz, gz, 0);
#endif
    __builtin_amdgcn_s_wait_tensorcnt(0);
  }
#else
  {
    int r = tid >> 5, seg = tid & 31;  // 32 threads per row, 64B each
    const uint4* src = (const uint4*)(qn + ((size_t)b * SS + qbase + r) * HH + seg * 32);
    uint4* dst = (uint4*)(&sQP[r * 1024 + seg * 32]);
#pragma unroll
    for (int i = 0; i < 4; ++i) dst[i] = src[i];
  }
#endif
  __syncthreads();

  // ---- Phase A: scores (in registers), ping-pong over k --------------------
  v8f acc[4];
#pragma unroll
  for (int c = 0; c < 4; ++c) acc[c] = zero_v8f();
  const int mbase0 = wave * 64;
  const __bf16* kbase = kn + ((size_t)b * MM + mbase0 + l16) * HH + half * 16;
  const size_t kcs = (size_t)16 * HH;
  const __bf16* aLds = &sQP[l16 * 1024 + half * 16];

  {
    v16bf a0 = *(const v16bf*)(aLds);
    v16bf k0[4];
#pragma unroll
    for (int c = 0; c < 4; ++c) k0[c] = *(const v16bf*)(kbase + c * kcs);
    int kk = 0;
    for (; kk + 64 < HH; kk += 64) {
      v16bf a1 = *(const v16bf*)(aLds + kk + 32);
      v16bf k1[4];
#pragma unroll
      for (int c = 0; c < 4; ++c) k1[c] = *(const v16bf*)(kbase + c * kcs + kk + 32);
#pragma unroll
      for (int c = 0; c < 4; ++c) acc[c] = wmma_bf16(a0, k0[c], acc[c]);
      a0 = *(const v16bf*)(aLds + kk + 64);
#pragma unroll
      for (int c = 0; c < 4; ++c) k0[c] = *(const v16bf*)(kbase + c * kcs + kk + 64);
#pragma unroll
      for (int c = 0; c < 4; ++c) acc[c] = wmma_bf16(a1, k1[c], acc[c]);
    }
    v16bf a1 = *(const v16bf*)(aLds + HH - 32);
    v16bf k1[4];
#pragma unroll
    for (int c = 0; c < 4; ++c) k1[c] = *(const v16bf*)(kbase + c * kcs + HH - 32);
#pragma unroll
    for (int c = 0; c < 4; ++c) acc[c] = wmma_bf16(a0, k0[c], acc[c]);
#pragma unroll
    for (int c = 0; c < 4; ++c) acc[c] = wmma_bf16(a1, k1[c], acc[c]);
  }

  // ---- Softmax over M=1024 -------------------------------------------------
  float rmax[8];
#pragma unroll
  for (int r = 0; r < 8; ++r) {
    rmax[r] = -3.402823466e38f;
#pragma unroll
    for (int c = 0; c < 4; ++c) rmax[r] = fmaxf(rmax[r], acc[c][r]);
  }
#pragma unroll
  for (int m = 1; m < 16; m <<= 1)
#pragma unroll
    for (int r = 0; r < 8; ++r) rmax[r] = fmaxf(rmax[r], __shfl_xor(rmax[r], m, 32));
  if (l16 == 0) {
#pragma unroll
    for (int r = 0; r < 8; ++r) wred[r + 8 * half][wave] = rmax[r];
  }
  __syncthreads();
  if (tid < 16) {
    float v = wred[tid][0];
#pragma unroll
    for (int w = 1; w < 16; ++w) v = fmaxf(v, wred[tid][w]);
    rowstat[tid] = v;
  }
  __syncthreads();
  float rmaxf[8];
#pragma unroll
  for (int r = 0; r < 8; ++r) rmaxf[r] = rowstat[r + 8 * half];

  float rsum[8];
#pragma unroll
  for (int r = 0; r < 8; ++r) rsum[r] = 0.0f;
#pragma unroll
  for (int c = 0; c < 4; ++c)
#pragma unroll
    for (int r = 0; r < 8; ++r) {
      float e = __expf(acc[c][r] - rmaxf[r]);  // TEMP == 1.0
      acc[c][r] = e;
      rsum[r] += e;
    }
#pragma unroll
  for (int m = 1; m < 16; m <<= 1)
#pragma unroll
    for (int r = 0; r < 8; ++r) rsum[r] += __shfl_xor(rsum[r], m, 32);
  if (l16 == 0) {
#pragma unroll
    for (int r = 0; r < 8; ++r) wred[r + 8 * half][wave] = rsum[r];
  }
  __syncthreads();
  if (tid < 16) {
    float v = 0.0f;
#pragma unroll
    for (int w = 0; w < 16; ++w) v += wred[tid][w];
    rowstat[tid] = v;
  }
  __syncthreads();

  // Write probabilities (bf16) into sQP, reusing the qn staging buffer.
  float rinv[8];
#pragma unroll
  for (int r = 0; r < 8; ++r) rinv[r] = 1.0f / rowstat[r + 8 * half];
#pragma unroll
  for (int c = 0; c < 4; ++c)
#pragma unroll
    for (int r = 0; r < 8; ++r)
      sQP[(r + 8 * half) * 1024 + mbase0 + c * 16 + l16] =
          (__bf16)(acc[c][r] * rinv[r]);
  __syncthreads();

  // ---- Phase C: retrieved = probs . mv  (B fragments from mv^T) ------------
  v8f outv[4];
#pragma unroll
  for (int c = 0; c < 4; ++c) outv[c] = zero_v8f();
  const int hbase0 = wave * 64;
  const __bf16* vbase = mvT + ((size_t)b * HH + hbase0 + l16) * (size_t)MM + half * 16;
  const size_t vcs = (size_t)16 * MM;

  {
    v16bf a0 = *(const v16bf*)(aLds);
    v16bf k0[4];
#pragma unroll
    for (int c = 0; c < 4; ++c) k0[c] = *(const v16bf*)(vbase + c * vcs);
    int kk = 0;
    for (; kk + 64 < MM; kk += 64) {
      v16bf a1 = *(const v16bf*)(aLds + kk + 32);
      v16bf k1[4];
#pragma unroll
      for (int c = 0; c < 4; ++c) k1[c] = *(const v16bf*)(vbase + c * vcs + kk + 32);
#pragma unroll
      for (int c = 0; c < 4; ++c) outv[c] = wmma_bf16(a0, k0[c], outv[c]);
      a0 = *(const v16bf*)(aLds + kk + 64);
#pragma unroll
      for (int c = 0; c < 4; ++c) k0[c] = *(const v16bf*)(vbase + c * vcs + kk + 64);
#pragma unroll
      for (int c = 0; c < 4; ++c) outv[c] = wmma_bf16(a1, k1[c], outv[c]);
    }
    v16bf a1 = *(const v16bf*)(aLds + MM - 32);
    v16bf k1[4];
#pragma unroll
    for (int c = 0; c < 4; ++c) k1[c] = *(const v16bf*)(vbase + c * vcs + MM - 32);
#pragma unroll
    for (int c = 0; c < 4; ++c) outv[c] = wmma_bf16(a0, k0[c], outv[c]);
#pragma unroll
    for (int c = 0; c < 4; ++c) outv[c] = wmma_bf16(a1, k1[c], outv[c]);
  }

#pragma unroll
  for (int c = 0; c < 4; ++c)
#pragma unroll
    for (int r = 0; r < 8; ++r)
      Rb[((size_t)b * SS + qbase + r + 8 * half) * HH + hbase0 + c * 16 + l16] =
          (__bf16)outv[c][r];
}

// ---------------------------------------------------------------------------
// Host-side orchestration.
// Inputs: 0 hidden_states (B,S,H) f32; 1 Wq (H,H); 2 bq; 3 Wk; 4 bk; 5 Wv;
//         6 bv; 7 Wo (H,2H); 8 bo.  Output: enhanced (B,S,H) f32.
// ---------------------------------------------------------------------------
extern "C" void kernel_launch(void* const* d_in, const int* in_sizes, int n_in,
                              void* d_out, int out_size, void* d_ws, size_t ws_size,
                              hipStream_t stream) {
  (void)in_sizes; (void)n_in; (void)out_size; (void)ws_size;

  const float* hs = (const float*)d_in[0];
  const float* Wq = (const float*)d_in[1];
  const float* bq = (const float*)d_in[2];
  const float* Wk = (const float*)d_in[3];
  const float* bk = (const float*)d_in[4];
  const float* Wv = (const float*)d_in[5];
  const float* bv = (const float*)d_in[6];
  const float* Wo = (const float*)d_in[7];
  const float* bo = (const float*)d_in[8];
  float* out = (float*)d_out;

  const size_t nBSH = (size_t)BB * SS * HH;  // 33,554,432
  const size_t nBMH = (size_t)BB * MM * HH;  // 8,388,608
  const size_t nHH  = (size_t)HH * HH;       // 1,048,576

  // Workspace partition (256B aligned).
  char* ws = (char*)d_ws;
  auto alloc = [&](size_t bytes) -> void* {
    void* p = (void*)ws;
    ws += (bytes + 255) & ~(size_t)255;
    return p;
  };
  __bf16* Xs   = (__bf16*)alloc(nBSH * 2);  // hidden_states bf16
  __bf16* Xm   = (__bf16*)alloc(nBMH * 2);  // gathered init_states bf16
  __bf16* Wqb  = (__bf16*)alloc(nHH * 2);
  __bf16* Wkb  = (__bf16*)alloc(nHH * 2);
  __bf16* Wvb  = (__bf16*)alloc(nHH * 2);
  __bf16* Wo1b = (__bf16*)alloc(nHH * 2);
  __bf16* Wo2b = (__bf16*)alloc(nHH * 2);
  float*  qk   = (float*)alloc(nBSH * 4);   // query_keys f32
  __bf16* qn   = (__bf16*)alloc(nBSH * 2);  // normalized queries bf16
  float*  mk   = (float*)alloc(nBMH * 4);   // memory_keys f32
  __bf16* kn   = (__bf16*)alloc(nBMH * 2);  // normalized keys bf16
  __bf16* mvT  = (__bf16*)alloc(nBMH * 2);  // memory_values^T bf16 (B,H,M)
  __bf16* Rb   = (__bf16*)alloc(nBSH * 2);  // retrieved bf16

  // 1) Converts / gather.
  f32_to_bf16_kernel<<<(nBSH + 255) / 256, 256, 0, stream>>>(hs, Xs, nBSH);
  f32_to_bf16_kernel<<<(nHH + 255) / 256, 256, 0, stream>>>(Wq, Wqb, nHH);
  f32_to_bf16_kernel<<<(nHH + 255) / 256, 256, 0, stream>>>(Wk, Wkb, nHH);
  f32_to_bf16_kernel<<<(nHH + 255) / 256, 256, 0, stream>>>(Wv, Wvb, nHH);
  split_wo_kernel<<<(nHH + 255) / 256, 256, 0, stream>>>(Wo, Wo1b, Wo2b);
  gather_bf16_kernel<<<(nBMH + 255) / 256, 256, 0, stream>>>(hs, Xm);

  // 2) memory_keys = Xm @ Wk^T + bk  (f32)
  gemm_bf16_wmma_kernel<<<dim3(HH / 512, MM / 16, BB), 256, 0, stream>>>(
      Xm, (long long)MM * HH, HH, Wkb, 0, HH, bk,
      mk, (long long)MM * HH, HH, nullptr, 0, 0, 0, HH, 0);

  // 3) memory_values^T = (Xm @ Wv^T + bv)^T  (bf16, transposed store)
  gemm_bf16_wmma_kernel<<<dim3(HH / 512, MM / 16, BB), 256, 0, stream>>>(
      Xm, (long long)MM * HH, HH, Wvb, 0, HH, bv,
      nullptr, 0, 0, mvT, (long long)HH * MM, MM, 1, HH, 0);

  // 4) query_keys = Xs @ Wq^T + bq  (f32)
  gemm_bf16_wmma_kernel<<<dim3(HH / 512, SS / 16, BB), 256, 0, stream>>>(
      Xs, (long long)SS * HH, HH, Wqb, 0, HH, bq,
      qk, (long long)SS * HH, HH, nullptr, 0, 0, 0, HH, 0);

  // 5) L2 normalize -> bf16.
  l2norm_bf16_kernel<<<(unsigned)(BB * SS), 256, 0, stream>>>(qk, qn);
  l2norm_bf16_kernel<<<(unsigned)(BB * MM), 256, 0, stream>>>(mk, kn);

  // 6) Fused attention (scores -> softmax -> AV), retrieved in bf16.
  attn_kernel<<<dim3(SS / 16, BB), 512, 0, stream>>>(qn, kn, mvT, Rb);

  // 7) enhanced = Xs @ Wo[:, :H]^T + bo ...
  gemm_bf16_wmma_kernel<<<dim3(HH / 512, SS / 16, BB), 256, 0, stream>>>(
      Xs, (long long)SS * HH, HH, Wo1b, 0, HH, bo,
      out, (long long)SS * HH, HH, nullptr, 0, 0, 0, HH, 0);

  // 8) ... += retrieved @ Wo[:, H:]^T  (accumulate into d_out)
  gemm_bf16_wmma_kernel<<<dim3(HH / 512, SS / 16, BB), 256, 0, stream>>>(
      Rb, (long long)SS * HH, HH, Wo2b, 0, HH, nullptr,
      out, (long long)SS * HH, HH, nullptr, 0, 0, 0, HH, 1);
}
